// MultiHeadAttention_30245159698594
// MI455X (gfx1250) — matrix operand order repaired and checked by reading, not compiled, not run on hardware
//
#include <hip/hip_runtime.h>
#include <hip/hip_bf16.h>
#include <math.h>

// ---------------- types ----------------
typedef __attribute__((ext_vector_type(16))) __bf16 v16bf;
typedef __attribute__((ext_vector_type(8)))  __bf16 v8bf;
typedef __attribute__((ext_vector_type(8)))  float  v8f;

#define BS 4
#define LX 2048
#define DM 1024
#define NH 16
#define DK 64
#define NEG_INF (-1e30f)

__device__ __forceinline__ v8f wmma_bf16(v16bf a, v16bf b, v8f c) {
  // D = A(16x32 bf16) * B(32x16 bf16) + C(16x16 f32)
  return __builtin_amdgcn_wmma_f32_16x16x32_bf16(false, a, false, b, (short)0, c,
                                                 false, false);
}

__device__ __forceinline__ v16bf pack16(v8bf lo, v8bf hi) {
  v16bf r;
#pragma unroll
  for (int i = 0; i < 8; ++i) { r[i] = lo[i]; r[i + 8] = hi[i]; }
  return r;
}

// A fragment: 16(M) x 32(K) bf16 from row-major A with leading dim lda.
// ISA layout: a[e] = A[row0 + lane%16][k0 + (e/8)*16 + (lane/16)*8 + e%8]
__device__ __forceinline__ v16bf load_afrag(const __bf16* A, int lda, int row0,
                                            int k0, int lane) {
  int m = lane & 15, h = lane >> 4;
  const __bf16* p = A + (size_t)(row0 + m) * lda + k0 + h * 8;
  return pack16(*(const v8bf*)p, *(const v8bf*)(p + 16));
}

// B fragment: 32(K) x 16(N) bf16, sourced from B^T stored [N][K] row-major.
// ISA layout: b[e] = B[k0 + (e/8)*16 + (lane/16)*8 + e%8][n0 + lane%16]
__device__ __forceinline__ v16bf load_bfrag(const __bf16* Bt, int ldb, int n0,
                                            int k0, int lane) {
  int n = lane & 15, h = lane >> 4;
  const __bf16* p = Bt + (size_t)(n0 + n) * ldb + k0 + h * 8;
  return pack16(*(const v8bf*)p, *(const v8bf*)(p + 16));
}

// Ping-pong double-buffered 16x64 GEMM tile over K=DM (no register rotation copies):
// stage 0 computes on {a0,b0} while {a1,b1} loads; stage 1 computes on {a1,b1}
// while {a0,b0} reloads for the next iteration.
__device__ __forceinline__ void gemm_tile_16x64(const __bf16* A, const __bf16* Bt,
                                                int row0, int n0, int lane,
                                                v8f acc[4]) {
  v16bf a0 = load_afrag(A, DM, row0, 0, lane);
  v16bf b0[4];
#pragma unroll
  for (int j = 0; j < 4; ++j) b0[j] = load_bfrag(Bt, DM, n0 + j * 16, 0, lane);

  for (int k0 = 0; k0 < DM; k0 += 64) {
    // prefetch k0+32 while computing k0
    v16bf a1 = load_afrag(A, DM, row0, k0 + 32, lane);
    v16bf b1[4];
#pragma unroll
    for (int j = 0; j < 4; ++j)
      b1[j] = load_bfrag(Bt, DM, n0 + j * 16, k0 + 32, lane);
#pragma unroll
    for (int j = 0; j < 4; ++j) acc[j] = wmma_bf16(a0, b0[j], acc[j]);
    // prefetch k0+64 (if any) while computing k0+32
    if (k0 + 64 < DM) {
      a0 = load_afrag(A, DM, row0, k0 + 64, lane);
#pragma unroll
      for (int j = 0; j < 4; ++j)
        b0[j] = load_bfrag(Bt, DM, n0 + j * 16, k0 + 64, lane);
    }
#pragma unroll
    for (int j = 0; j < 4; ++j) acc[j] = wmma_bf16(a1, b1[j], acc[j]);
  }
}

// ---------------- kernel 1: x [b][d][l] (f32) -> xt [b][l][d] (bf16) ----------------
// Tiled LDS transpose: coalesced reads along l, coalesced writes along d.
// grid: (LX/32, DM/32, BS), block (32, 8)
__global__ void __launch_bounds__(256) k_transpose_x(const float* __restrict__ x,
                                                     __hip_bfloat16* __restrict__ xt_) {
  __bf16* xt = reinterpret_cast<__bf16*>(xt_);
  __shared__ float tile[32][33];
  int l0 = blockIdx.x * 32, d0 = blockIdx.y * 32, b = blockIdx.z;
  int tx = threadIdx.x, ty = threadIdx.y;
#pragma unroll
  for (int i = 0; i < 4; ++i) {
    int d = d0 + ty + i * 8;
    tile[ty + i * 8][tx] = x[((size_t)b * DM + d) * LX + l0 + tx];
  }
  __syncthreads();
#pragma unroll
  for (int i = 0; i < 4; ++i) {
    int l = l0 + ty + i * 8;
    xt[((size_t)b * LX + l) * DM + d0 + tx] = (__bf16)tile[tx][ty + i * 8];
  }
}

// ---------------- kernel 2: W [k][n] (f32) -> Wt [n][k] (bf16) ----------------
// grid: (DM/32, DM/32), block (32, 8)
__global__ void __launch_bounds__(256) k_transpose_w(const float* __restrict__ W,
                                                     __hip_bfloat16* __restrict__ Wt_) {
  __bf16* Wt = reinterpret_cast<__bf16*>(Wt_);
  __shared__ float tile[32][33];
  int k0 = blockIdx.x * 32, n0 = blockIdx.y * 32;
  int tx = threadIdx.x, ty = threadIdx.y;
#pragma unroll
  for (int i = 0; i < 4; ++i)
    tile[ty + i * 8][tx] = W[((size_t)(k0 + ty + i * 8)) * DM + n0 + tx];
  __syncthreads();
#pragma unroll
  for (int i = 0; i < 4; ++i)
    Wt[((size_t)(n0 + ty + i * 8)) * DM + k0 + tx] = (__bf16)tile[tx][ty + i * 8];
}

// ---------------- kernel 3: fused QKV projection ----------------
// grid: (256, BS, 3). Each wave computes a 16(l) x 64(d_out) tile (one head's d range).
__global__ void __launch_bounds__(256) k_proj_qkv(
    const __hip_bfloat16* __restrict__ xt_,
    const __hip_bfloat16* __restrict__ Wqt_, const __hip_bfloat16* __restrict__ Wkt_,
    const __hip_bfloat16* __restrict__ Wvt_,
    const float* __restrict__ bq, const float* __restrict__ bk,
    const float* __restrict__ bv,
    __hip_bfloat16* __restrict__ Q_, __hip_bfloat16* __restrict__ K_,
    __hip_bfloat16* __restrict__ Vt_) {
  const __bf16* xt = reinterpret_cast<const __bf16*>(xt_);
  int b = blockIdx.y;
  int which = blockIdx.z;  // 0=Q 1=K 2=V  (uniform per block)
  const __bf16* W = reinterpret_cast<const __bf16*>(which == 0 ? Wqt_
                                                   : which == 1 ? Wkt_ : Wvt_);
  const float* bias = (which == 0) ? bq : (which == 1) ? bk : bv;

  int wave = blockIdx.x * (blockDim.x >> 5) + (threadIdx.x >> 5);  // 0..2047
  int lane = threadIdx.x & 31;
  int mt = wave >> 4;   // 0..127
  int nt = wave & 15;   // 0..15 -> head
  int row0 = mt * 16;
  int n0 = nt * 64;
  const __bf16* A = xt + (size_t)b * LX * DM;

  v8f acc[4];
#pragma unroll
  for (int j = 0; j < 4; ++j) acc[j] = (v8f){0, 0, 0, 0, 0, 0, 0, 0};
  gemm_tile_16x64(A, W, row0, n0, lane, acc);

  int m = lane & 15, h = lane >> 4;
  __bf16* Q = reinterpret_cast<__bf16*>(Q_);
  __bf16* K = reinterpret_cast<__bf16*>(K_);
  __bf16* Vt = reinterpret_cast<__bf16*>(Vt_);
#pragma unroll
  for (int j = 0; j < 4; ++j) {
    int dcol = j * 16 + m;                 // 0..63 within head `nt`
    float bv_ = bias[n0 + dcol];
#pragma unroll
    for (int r = 0; r < 8; ++r) {
      int row = row0 + r + 8 * h;          // l index
      __bf16 o = (__bf16)(acc[j][r] + bv_);
      if (which == 0)
        Q[(((size_t)b * NH + nt) * LX + row) * DK + dcol] = o;
      else if (which == 1)
        K[(((size_t)b * NH + nt) * LX + row) * DK + dcol] = o;
      else
        Vt[(((size_t)b * NH + nt) * DK + dcol) * LX + row] = o;  // V transposed
    }
  }
}

// ---------------- kernel 4: flash attention per (b, head) ----------------
// grid: (LX/128, NH, BS); block 256 = 8 waves; each wave owns 16 q rows.
// K fragments double-buffered; V fragments issued before the softmax VALU block.
__global__ void __launch_bounds__(256) k_attention(
    const __hip_bfloat16* __restrict__ Q_, const __hip_bfloat16* __restrict__ K_,
    const __hip_bfloat16* __restrict__ Vt_, const int* __restrict__ mask,
    __hip_bfloat16* __restrict__ ctx_) {
  const __bf16* Q = reinterpret_cast<const __bf16*>(Q_);
  const __bf16* K = reinterpret_cast<const __bf16*>(K_);
  const __bf16* Vt = reinterpret_cast<const __bf16*>(Vt_);
  __bf16* ctx = reinterpret_cast<__bf16*>(ctx_);

  int b = blockIdx.z, hd = blockIdx.y;
  int wid = threadIdx.x >> 5, lane = threadIdx.x & 31;
  int m = lane & 15, h = lane >> 4;
  int q0 = blockIdx.x * 128 + wid * 16;

  const __bf16* Qb = Q + ((size_t)b * NH + hd) * LX * DK;
  const __bf16* Kb = K + ((size_t)b * NH + hd) * LX * DK;
  const __bf16* Vb = Vt + ((size_t)b * NH + hd) * DK * LX;  // [d][l]
  const int* mk = mask + (size_t)b * LX;

  // per-wave P tile: 16 q rows x 32 keys (row stride 40 keeps 16B alignment)
  __shared__ __align__(16) __bf16 pt[8][16][40];

  // Q fragments for this wave's 16 rows (d = 0..63 in two K-steps)
  v16bf qa0 = load_afrag(Qb, DK, q0, 0, lane);
  v16bf qa1 = load_afrag(Qb, DK, q0, 32, lane);

  v8f acc[4];
#pragma unroll
  for (int j = 0; j < 4; ++j) acc[j] = (v8f){0, 0, 0, 0, 0, 0, 0, 0};
  float mrun[8], srun[8];
#pragma unroll
  for (int r = 0; r < 8; ++r) { mrun[r] = -INFINITY; srun[r] = 0.0f; }

  const float scale = 0.125f;  // 1/sqrt(64)
  const int nC = LX / 32;

  // preload K fragments for chunk 0:  kf[tile][kstep]
  v16bf kf[2][2];
#pragma unroll
  for (int t = 0; t < 2; ++t) {
    kf[t][0] = load_bfrag(Kb, DK, t * 16, 0, lane);
    kf[t][1] = load_bfrag(Kb, DK, t * 16, 32, lane);
  }

  for (int c = 0; c < nC; ++c) {
    int kb0 = c * 32;
    // ---- S = Q K^T for two 16-key tiles (fragments loaded last iteration) ----
    v8f S[2];
#pragma unroll
    for (int t = 0; t < 2; ++t) {
      S[t] = (v8f){0, 0, 0, 0, 0, 0, 0, 0};
      S[t] = wmma_bf16(qa0, kf[t][0], S[t]);
      S[t] = wmma_bf16(qa1, kf[t][1], S[t]);
    }
    // ---- issue V loads for this chunk (consumed after softmax) ----
    v16bf vf[4];
#pragma unroll
    for (int j = 0; j < 4; ++j) vf[j] = load_bfrag(Vb, LX, j * 16, kb0, lane);
    // ---- issue K loads for next chunk (hidden under softmax + P@V) ----
    if (c + 1 < nC) {
      int kn = kb0 + 32;
#pragma unroll
      for (int t = 0; t < 2; ++t) {
        kf[t][0] = load_bfrag(Kb, DK, kn + t * 16, 0, lane);
        kf[t][1] = load_bfrag(Kb, DK, kn + t * 16, 32, lane);
      }
    }
    // ---- mask + scale (each lane's fragment column = one key) ----
    float sc[2][8];
#pragma unroll
    for (int t = 0; t < 2; ++t) {
      int key = kb0 + t * 16 + m;
      bool valid = (mk[key] != 0);
#pragma unroll
      for (int r = 0; r < 8; ++r)
        sc[t][r] = valid ? S[t][r] * scale : NEG_INF;
    }
    // ---- online softmax, per row r (rows r and r+8 tracked by lane halves) ----
#pragma unroll
    for (int r = 0; r < 8; ++r) {
      float v = fmaxf(sc[0][r], sc[1][r]);
#pragma unroll
      for (int o = 8; o >= 1; o >>= 1) v = fmaxf(v, __shfl_xor(v, o));
      float nm = fmaxf(mrun[r], v);
      float rescale = __expf(mrun[r] - nm);
      mrun[r] = nm;
      float psum = 0.0f;
#pragma unroll
      for (int t = 0; t < 2; ++t) {
        float p = __expf(sc[t][r] - nm);
        pt[wid][r + 8 * h][t * 16 + m] = (__bf16)p;  // wave-local LDS, no barrier
        psum += p;
      }
#pragma unroll
      for (int o = 8; o >= 1; o >>= 1) psum += __shfl_xor(psum, o);
      srun[r] = srun[r] * rescale + psum;
#pragma unroll
      for (int j = 0; j < 4; ++j) acc[j][r] *= rescale;
    }
    // ---- P (16x32) from LDS as A fragment ----
    const __bf16* pp = &pt[wid][m][h * 8];
    v16bf pa = pack16(*(const v8bf*)pp, *(const v8bf*)(pp + 16));
    // ---- acc += P * V  (V^T stored [d][l] -> contiguous B fragments) ----
#pragma unroll
    for (int j = 0; j < 4; ++j) acc[j] = wmma_bf16(pa, vf[j], acc[j]);
  }

  // ---- normalize + store ctx [b][l][h*64+d] (bf16) ----
#pragma unroll
  for (int r = 0; r < 8; ++r) {
    float inv = 1.0f / srun[r];
    int row = q0 + r + 8 * h;
#pragma unroll
    for (int j = 0; j < 4; ++j) {
      int col = hd * DK + j * 16 + m;
      ctx[((size_t)b * LX + row) * DM + col] = (__bf16)(acc[j][r] * inv);
    }
  }
}

// ---------------- kernel 5: out = ctx @ Wf + bf, store transposed fp32 ----------------
// grid: (256, BS); wave computes 16(l) x 64(n) tile.
__global__ void __launch_bounds__(256) k_out_proj(
    const __hip_bfloat16* __restrict__ ctx_, const __hip_bfloat16* __restrict__ Wft_,
    const float* __restrict__ bf_, float* __restrict__ out) {
  const __bf16* ctx = reinterpret_cast<const __bf16*>(ctx_);
  const __bf16* Wft = reinterpret_cast<const __bf16*>(Wft_);
  int b = blockIdx.y;
  int wave = blockIdx.x * (blockDim.x >> 5) + (threadIdx.x >> 5);
  int lane = threadIdx.x & 31;
  int mt = wave >> 4, nt = wave & 15;
  int row0 = mt * 16, n0 = nt * 64;
  const __bf16* A = ctx + (size_t)b * LX * DM;

  v8f acc[4];
#pragma unroll
  for (int j = 0; j < 4; ++j) acc[j] = (v8f){0, 0, 0, 0, 0, 0, 0, 0};
  gemm_tile_16x64(A, Wft, row0, n0, lane, acc);

  int m = lane & 15, h = lane >> 4;
#pragma unroll
  for (int j = 0; j < 4; ++j) {
    int col = n0 + j * 16 + m;       // d_model index
    float bv = bf_[col];
#pragma unroll
    for (int r = 0; r < 8; ++r) {
      int row = row0 + r + 8 * h;    // l index
      out[((size_t)b * DM + col) * LX + row] = acc[j][r] + bv;  // [b][d][l]
    }
  }
}

// ---------------- launcher ----------------
extern "C" void kernel_launch(void* const* d_in, const int* in_sizes, int n_in,
                              void* d_out, int out_size, void* d_ws, size_t ws_size,
                              hipStream_t stream) {
  (void)in_sizes; (void)n_in; (void)out_size; (void)ws_size;
  const float* x  = (const float*)d_in[0];
  const int*   mask = (const int*)d_in[1];
  const float* Wq = (const float*)d_in[2];
  const float* bq = (const float*)d_in[3];
  const float* Wk = (const float*)d_in[4];
  const float* bk = (const float*)d_in[5];
  const float* Wv = (const float*)d_in[6];
  const float* bv = (const float*)d_in[7];
  const float* Wf = (const float*)d_in[8];
  const float* bf = (const float*)d_in[9];

  // workspace layout (bf16 buffers), all 256B aligned
  char* ws = (char*)d_ws;
  const size_t SZ_XT  = (size_t)BS * LX * DM * 2;       // 16 MB
  const size_t SZ_W   = (size_t)DM * DM * 2;            // 2 MB each
  const size_t SZ_QKV = (size_t)BS * NH * LX * DK * 2;  // 16 MB each
  __hip_bfloat16* xt  = (__hip_bfloat16*)(ws);
  __hip_bfloat16* Wqt = (__hip_bfloat16*)(ws + SZ_XT);
  __hip_bfloat16* Wkt = (__hip_bfloat16*)(ws + SZ_XT + SZ_W);
  __hip_bfloat16* Wvt = (__hip_bfloat16*)(ws + SZ_XT + 2 * SZ_W);
  __hip_bfloat16* Wft = (__hip_bfloat16*)(ws + SZ_XT + 3 * SZ_W);
  __hip_bfloat16* Qb  = (__hip_bfloat16*)(ws + SZ_XT + 4 * SZ_W);
  __hip_bfloat16* Kb  = (__hip_bfloat16*)(ws + SZ_XT + 4 * SZ_W + SZ_QKV);
  __hip_bfloat16* Vtb = (__hip_bfloat16*)(ws + SZ_XT + 4 * SZ_W + 2 * SZ_QKV);
  __hip_bfloat16* ctx = (__hip_bfloat16*)(ws + SZ_XT + 4 * SZ_W + 3 * SZ_QKV);
  // total = 16 + 8 + 4*16 = 88 MB

  // 1) transpose+convert x (tiled)
  {
    dim3 grid(LX / 32, DM / 32, BS), blk(32, 8);
    k_transpose_x<<<grid, blk, 0, stream>>>(x, xt);
  }
  // 2) transpose+convert weights (tiled)
  {
    dim3 grid(DM / 32, DM / 32), blk(32, 8);
    k_transpose_w<<<grid, blk, 0, stream>>>(Wq, Wqt);
    k_transpose_w<<<grid, blk, 0, stream>>>(Wk, Wkt);
    k_transpose_w<<<grid, blk, 0, stream>>>(Wv, Wvt);
    k_transpose_w<<<grid, blk, 0, stream>>>(Wf, Wft);
  }
  // 3) QKV projections
  {
    dim3 grid(256, BS, 3);
    k_proj_qkv<<<grid, 256, 0, stream>>>(xt, Wqt, Wkt, Wvt, bq, bk, bv, Qb, Kb, Vtb);
  }
  // 4) attention
  {
    dim3 grid(LX / 128, NH, BS);
    k_attention<<<grid, 256, 0, stream>>>(Qb, Kb, Vtb, mask, ctx);
  }
  // 5) output projection (+ fused transpose to [b][d_model][l] fp32)
  {
    dim3 grid(256, BS);
    k_out_proj<<<grid, 256, 0, stream>>>(ctx, Wft, bf, (float*)d_out);
  }
}
